// MixLogAttnCoupling_86414741995536
// MI455X (gfx1250) — matrix operand order repaired and checked by reading, not compiled, not run on hardware
//
#include <hip/hip_runtime.h>
#include <hip/hip_bf16.h>
#include <stdint.h>

typedef __attribute__((ext_vector_type(16))) _Float16 v16h;
typedef __attribute__((ext_vector_type(8)))  _Float16 v8h;
typedef __attribute__((ext_vector_type(8)))  float    v8f;

static constexpr int NB   = 64;     // batch
static constexpr int TOK  = 1024;   // tokens per batch (32x32)
static constexpr int FC   = 96;     // feature channels

// ---------------------------------------------------------------------------
// WMMA helpers (contiguous k order k = 16*(lane/16) + i, same for A and B)
// ---------------------------------------------------------------------------
__device__ __forceinline__ v8f wmma16(v16h a, v16h b, v8f c) {
  return __builtin_amdgcn_wmma_f32_16x16x32_f16(false, a, false, b, (short)0, c,
                                                false, false);
}

// A fragment 16(M) x 32(K) f16 from row-major src (rows = M, K contiguous, ld halves)
// lane: M = lane&15 ; lanes 0-15 hold K 0-15 ; lanes 16-31 hold K 16-31
__device__ __forceinline__ v16h load_a_frag(const _Float16* p0, int ld, int lane) {
  const _Float16* p = p0 + (size_t)(lane & 15) * ld + ((lane & 16) ? 16 : 0);
  v8h lo = *(const v8h*)(p);
  v8h hi = *(const v8h*)(p + 8);
  v16h a;
#pragma unroll
  for (int i = 0; i < 8; ++i) { a[i] = lo[i]; a[i + 8] = hi[i]; }
  return a;
}

// B fragment 32(K) x 16(N) f16 gathered from row-major [K][ld] source.
// lane: N = lane&15 ; lanes 0-15 hold K=0..15, lanes 16-31 hold K=16..31
__device__ __forceinline__ v16h load_b_rm(const _Float16* p0, int ld, int lane) {
  const _Float16* p = p0 + (lane & 15) + ((lane & 16) ? 16 * (size_t)ld : 0);
  v16h b;
#pragma unroll
  for (int i = 0; i < 16; ++i) b[i] = p[(size_t)i * ld];
  return b;
}

__device__ __forceinline__ float sigmf(float v) { return 1.f / (1.f + __expf(-v)); }
__device__ __forceinline__ float logsigf(float v) {
  return fminf(v, 0.f) - log1pf(__expf(-fabsf(v)));
}

// ---------------------------------------------------------------------------
// init: log-det output seed + LN stats zero
// ---------------------------------------------------------------------------
__global__ void init_kernel(const float* ld_in, float* outld, float* stats) {
  int i = threadIdx.x;
  if (i < NB) outld[i] = ld_in[i];
  if (i < 4 * NB) stats[i] = 0.f;   // [ln1 sum,sumsq]x64 + [ln2]x64
}

// ---------------------------------------------------------------------------
// Pack a row-major weight view W(K,N) (+ taps) into B-fragment-major f16 tiles.
// value(K,N,t) = src[N*sN + K*sK + t] (0 if N >= Nvalid).
// packed[( (t*Ktiles + kt)*Ntiles + nt )*512 + lane*16 + i]
//   with K = kt*32 + i + (lane>=16 ? 16 : 0), N = nt*16 + (lane&15)
// ---------------------------------------------------------------------------
__global__ void pack_kernel(const float* src, _Float16* dst, int Ktiles, int Ntiles,
                            int Nvalid, int sN, int sK, int taps) {
  size_t total = (size_t)taps * Ktiles * Ntiles * 512;
  for (size_t idx = (size_t)blockIdx.x * 256 + threadIdx.x; idx < total;
       idx += (size_t)gridDim.x * 256) {
    int li = (int)(idx & 511);
    size_t tile = idx >> 9;
    int nt = (int)(tile % Ntiles);
    size_t tmp = tile / Ntiles;
    int kt = (int)(tmp % Ktiles);
    int t  = (int)(tmp / Ktiles);
    int lane = li >> 4, i = li & 15;
    int K = kt * 32 + i + ((lane & 16) ? 16 : 0);
    int N = nt * 16 + (lane & 15);
    float v = 0.f;
    if (N < Nvalid) v = src[(size_t)N * sN + (size_t)K * sK + t];
    dst[idx] = (_Float16)v;
  }
}

// ---------------------------------------------------------------------------
// checker split: z (64,3,64,64) -> z0, z1 (64,6,32,32)
// ---------------------------------------------------------------------------
__global__ void split_kernel(const float* z, float* z0, float* z1) {
  size_t idx = (size_t)blockIdx.x * 256 + threadIdx.x;  // 64*3*1024 threads
  int b = (int)(idx / (3 * 1024));
  int rem = (int)(idx % (3 * 1024));
  int c = rem / 1024, t = rem % 1024;
  int yh = t >> 5, xh = t & 31;
  const float* zp = z + ((size_t)b * 3 + c) * 4096;
  float p00 = zp[(2 * yh) * 64 + 2 * xh];
  float p01 = zp[(2 * yh) * 64 + 2 * xh + 1];
  float p10 = zp[(2 * yh + 1) * 64 + 2 * xh];
  float p11 = zp[(2 * yh + 1) * 64 + 2 * xh + 1];
  size_t base = (size_t)b * 6144 + t;
  z0[base + (size_t)c * 1024]       = p00;
  z0[base + (size_t)(c + 3) * 1024] = p11;
  z1[base + (size_t)c * 1024]       = p01;
  z1[base + (size_t)(c + 3) * 1024] = p10;
}

// ---------------------------------------------------------------------------
// conv1: z1 (b,6,32,32) -> x token-major [b][1024][96] f32 + f16 (direct; tiny)
// ---------------------------------------------------------------------------
__global__ void conv1_kernel(const float* z1, const float* w, const float* bias,
                             float* xf, _Float16* xh) {
  size_t idx = (size_t)blockIdx.x * 256 + threadIdx.x;  // 1024*96 per batch
  int b = blockIdx.y;
  int t = (int)(idx / FC), c = (int)(idx % FC);
  int y = t >> 5, x = t & 31;
  float acc = bias[c];
  const float* zb = z1 + (size_t)b * 6 * 1024;
  const float* wc = w + (size_t)c * 54;
#pragma unroll
  for (int ci = 0; ci < 6; ++ci)
#pragma unroll
    for (int dy = 0; dy < 3; ++dy) {
      int yy = y + dy - 1;
      if (yy < 0 || yy > 31) continue;
#pragma unroll
      for (int dx = 0; dx < 3; ++dx) {
        int xx = x + dx - 1;
        if (xx < 0 || xx > 31) continue;
        acc += zb[(size_t)ci * 1024 + yy * 32 + xx] * wc[ci * 9 + dy * 3 + dx];
      }
    }
  size_t o = ((size_t)b * TOK + t) * FC + c;
  xf[o] = acc;
  xh[o] = (_Float16)acc;
}

// ---------------------------------------------------------------------------
// 3x3 conv as 9 shifted 1x1 GEMMs via WMMA. In: token-major f16 [b][1024][96],
// packed taps, out f32 [b][1024][Npad]. block=128 (4 waves), each wave a 16x16 tile.
// grid: (Npad/32, 32 rows, 64 batch). LDS: 3 rows x 34 cols x 96ch halo tile.
// ---------------------------------------------------------------------------
__global__ void convtap_kernel(const _Float16* in, const _Float16* pk,
                               const float* bias, float* out, int Npad, int Ntiles,
                               int Nvalid) {
  __shared__ __align__(32) _Float16 tile[3 * 34 * FC];
  int b = blockIdx.z, row = blockIdx.y;
  const _Float16* src = in + (size_t)b * TOK * FC;
  for (int idx = threadIdx.x; idx < 3 * 34 * FC; idx += 128) {
    int dy = idx / (34 * FC);
    int rem = idx % (34 * FC);
    int col = rem / FC, ch = rem % FC;
    int y = row + dy - 1, xc = col - 1;
    _Float16 v = (_Float16)0.f;
    if (y >= 0 && y < 32 && xc >= 0 && xc < 32)
      v = src[((size_t)y * 32 + xc) * FC + ch];
    tile[idx] = v;
  }
  __syncthreads();

  int wave = threadIdx.x >> 5, lane = threadIdx.x & 31;
  int mt = wave & 1;                 // token half (16 tokens)
  int ntg = blockIdx.x * 2 + (wave >> 1);
  v8f acc = {};
#pragma unroll
  for (int t = 0; t < 9; ++t) {
    int dy = t / 3, dx = t % 3;
    const _Float16* ab = &tile[((size_t)dy * 34 + mt * 16 + dx) * FC];
    const _Float16* bb = pk + (size_t)t * 3 * Ntiles * 512;
#pragma unroll
    for (int ks = 0; ks < 3; ++ks) {
      v16h a = load_a_frag(ab + ks * 32, FC, lane);
      v16h bf = *(const v16h*)(bb + ((size_t)(ks * Ntiles + ntg)) * 512 + lane * 16);
      acc = wmma16(a, bf, acc);
    }
  }
  int col = ntg * 16 + (lane & 15);
  float bv = (col < Nvalid) ? bias[col] : 0.f;
  int rb = (lane & 16) ? 8 : 0;
  float* op = out + ((size_t)b * TOK + row * 32 + mt * 16) * Npad;
#pragma unroll
  for (int r = 0; r < 8; ++r) op[(size_t)(r + rb) * Npad + col] = acc[r] + bv;
}

// ---------------------------------------------------------------------------
// GEMM over K=96: A f16 token-major [b][1024][96] x packed B -> out (f32/f16)
// grid: (Ntiles/2, 32, 64), block 128
// ---------------------------------------------------------------------------
__global__ void gemm96_kernel(const _Float16* A, const _Float16* Bp, const float* bias,
                              int Ntiles, int Npad, int Nvalid, float* outf,
                              _Float16* outh) {
  int wave = threadIdx.x >> 5, lane = threadIdx.x & 31;
  int b = blockIdx.z, row = blockIdx.y;
  int mt = wave & 1;
  int ntg = blockIdx.x * 2 + (wave >> 1);
  const _Float16* ab = A + ((size_t)b * TOK + row * 32 + mt * 16) * FC;
  v8f acc = {};
#pragma unroll
  for (int ks = 0; ks < 3; ++ks) {
    v16h a = load_a_frag(ab + ks * 32, FC, lane);
    v16h bf = *(const v16h*)(Bp + ((size_t)(ks * Ntiles + ntg)) * 512 + lane * 16);
    acc = wmma16(a, bf, acc);
  }
  int col = ntg * 16 + (lane & 15);
  float bv = (col < Nvalid) ? bias[col] : 0.f;
  int rb = (lane & 16) ? 8 : 0;
#pragma unroll
  for (int r = 0; r < 8; ++r) {
    float v = acc[r] + bv;
    size_t o = ((size_t)b * TOK + row * 32 + mt * 16 + r + rb) * Npad + col;
    if (outf) outf[o] = v;
    if (outh) outh[o] = (_Float16)v;
  }
}

// ---------------------------------------------------------------------------
// Flash attention. qkvh f16 [b][1024][288] (q|k|v). One wave per 16-query tile.
// block 128 (4 waves), grid (16, 64).
// ---------------------------------------------------------------------------
__global__ void attn_kernel(const _Float16* qkvh, _Float16* oh) {
  __shared__ __align__(32) _Float16 smem[4 * 3584];
  int wave = threadIdx.x >> 5, lane = threadIdx.x & 31;
  _Float16* pst = &smem[wave * 3584];      // 16x32 P tile
  _Float16* vst = pst + 512;               // 32x96 V tile
  int b = blockIdx.y;
  int qbase = blockIdx.x * 64 + wave * 16;
  const _Float16* qk = qkvh + (size_t)b * TOK * 288;

  v16h qf[3];
  const _Float16* qrow = qk + (size_t)qbase * 288;
#pragma unroll
  for (int ks = 0; ks < 3; ++ks) qf[ks] = load_a_frag(qrow + ks * 32, 288, lane);

  float mrow[8], lrow[8];
  v8f o[6];
#pragma unroll
  for (int r = 0; r < 8; ++r) { mrow[r] = -1e30f; lrow[r] = 0.f; }
#pragma unroll
  for (int n = 0; n < 6; ++n) o[n] = (v8f){};
  const float scale = 0.1020620726159658f;  // 1/sqrt(96)
  int kb16 = (lane & 16) ? 16 : 0;

  for (int kc = 0; kc < 32; ++kc) {
    int kt = kc * 32;
    v8f s0 = {}, s1 = {};
#pragma unroll
    for (int ks = 0; ks < 3; ++ks) {
      const _Float16* kr0 =
          qk + (size_t)(kt + (lane & 15)) * 288 + FC + ks * 32 + kb16;
      const _Float16* kr1 =
          qk + (size_t)(kt + 16 + (lane & 15)) * 288 + FC + ks * 32 + kb16;
      v16h b0 = *(const v16h*)kr0;
      v16h b1 = *(const v16h*)kr1;
      s0 = wmma16(qf[ks], b0, s0);
      s1 = wmma16(qf[ks], b1, s1);
    }
    float a0[8], a1[8], rmax[8];
#pragma unroll
    for (int r = 0; r < 8; ++r) {
      a0[r] = s0[r] * scale;
      a1[r] = s1[r] * scale;
      rmax[r] = fmaxf(a0[r], a1[r]);
    }
#pragma unroll
    for (int off = 1; off < 16; off <<= 1)
#pragma unroll
      for (int r = 0; r < 8; ++r) rmax[r] = fmaxf(rmax[r], __shfl_xor(rmax[r], off, 32));
    float p0[8], p1[8], rs[8], al[8];
#pragma unroll
    for (int r = 0; r < 8; ++r) {
      float mn = fmaxf(mrow[r], rmax[r]);
      al[r] = __expf(mrow[r] - mn);
      p0[r] = __expf(a0[r] - mn);
      p1[r] = __expf(a1[r] - mn);
      rs[r] = p0[r] + p1[r];
      mrow[r] = mn;
    }
#pragma unroll
    for (int off = 1; off < 16; off <<= 1)
#pragma unroll
      for (int r = 0; r < 8; ++r) rs[r] += __shfl_xor(rs[r], off, 32);
#pragma unroll
    for (int r = 0; r < 8; ++r) lrow[r] = lrow[r] * al[r] + rs[r];
#pragma unroll
    for (int n = 0; n < 6; ++n)
#pragma unroll
      for (int r = 0; r < 8; ++r) o[n][r] *= al[r];

    // P (C-frag) -> LDS row-major 16x32
    int rb = (lane & 16) ? 8 : 0, c = lane & 15;
#pragma unroll
    for (int r = 0; r < 8; ++r) {
      pst[(r + rb) * 32 + c]      = (_Float16)p0[r];
      pst[(r + rb) * 32 + 16 + c] = (_Float16)p1[r];
    }
    // stage V chunk 32x96 into LDS (lane = key row)
    const _Float16* vr = qk + (size_t)(kt + lane) * 288 + 192;
#pragma unroll
    for (int i = 0; i < 6; ++i)
      *(v16h*)(vst + (size_t)lane * FC + i * 16) = *(const v16h*)(vr + i * 16);

    asm volatile("s_wait_dscnt 0" ::: "memory");

    v16h ap = load_a_frag(pst, 32, lane);
#pragma unroll
    for (int n = 0; n < 6; ++n) {
      v16h bv = load_b_rm(vst + n * 16, FC, lane);
      o[n] = wmma16(ap, bv, o[n]);
    }
  }
  int rb = (lane & 16) ? 8 : 0, c = lane & 15;
  _Float16* op = oh + ((size_t)b * TOK + qbase) * FC;
#pragma unroll
  for (int n = 0; n < 6; ++n)
#pragma unroll
    for (int r = 0; r < 8; ++r)
      op[(size_t)(r + rb) * FC + n * 16 + c] = (_Float16)(o[n][r] / lrow[r]);
}

// ---------------------------------------------------------------------------
// gate: out = xin + ga*sigmoid(gb) ; + per-batch sum/sumsq atomics (LN stats)
// ---------------------------------------------------------------------------
__global__ void gate_kernel(const float* xin, const float* g, float* xout,
                            float* stats) {
  size_t idx = (size_t)blockIdx.x * 256 + threadIdx.x;  // 1024*96 per batch
  int b = blockIdx.y;
  int t = (int)(idx / FC), c = (int)(idx % FC);
  const float* gp = g + ((size_t)b * TOK + t) * 192;
  float v = xin[((size_t)b * TOK + t) * FC + c] + gp[c] * sigmf(gp[c + FC]);
  xout[((size_t)b * TOK + t) * FC + c] = v;
  float s = v, s2 = v * v;
#pragma unroll
  for (int off = 16; off > 0; off >>= 1) {
    s += __shfl_xor(s, off, 32);
    s2 += __shfl_xor(s2, off, 32);
  }
  if ((threadIdx.x & 31) == 0) {
    atomicAdd(&stats[b * 2], s);
    atomicAdd(&stats[b * 2 + 1], s2);
  }
}

// ---------------------------------------------------------------------------
// layer norm apply: per-batch mean/var (stats), per-(c,y,x) gamma/beta
// ---------------------------------------------------------------------------
__global__ void ln_kernel(const float* xin, const float* stats, const float* gam,
                          const float* bet, float* outf, _Float16* outh) {
  size_t idx = (size_t)blockIdx.x * 256 + threadIdx.x;
  int b = blockIdx.y;
  int t = (int)(idx / FC), c = (int)(idx % FC);
  const float N = (float)(FC * TOK);
  float mean = stats[b * 2] / N;
  float var = stats[b * 2 + 1] / N - mean * mean;
  float rstd = rsqrtf(var + 1e-5f);
  size_t o = ((size_t)b * TOK + t) * FC + c;
  float v = (xin[o] - mean) * rstd;
  v = v * gam[(size_t)c * 1024 + t] + bet[(size_t)c * 1024 + t];
  if (outf) outf[o] = v;
  outh[o] = (_Float16)v;
}

// ---------------------------------------------------------------------------
// final: mixture-of-logistics transform + checker merge + log-det reduction
// params f32 token-major [b][1024][160] (valid 156)
// ---------------------------------------------------------------------------
__global__ void final_kernel(const float* z0, const float* z1, const float* params,
                             const float* a_ls, const float* a_bs, float* out,
                             float* outld) {
  size_t idx = (size_t)blockIdx.x * 256 + threadIdx.x;  // 64*6*1024
  int b = (int)(idx / 6144);
  int rem = (int)(idx % 6144);
  int c0 = rem / 1024, t = rem % 1024;
  const float* pp = params + ((size_t)b * TOK + t) * 160;
  float z0v = z0[(size_t)b * 6144 + (size_t)c0 * 1024 + t];
  float pa = tanhf(pp[c0]) * a_ls[0] + a_bs[0];
  float pb = pp[6 + c0];
  float lp[8];
  float mx = -1e30f;
#pragma unroll
  for (int k = 0; k < 8; ++k) { lp[k] = pp[12 + k * 6 + c0]; mx = fmaxf(mx, lp[k]); }
  float sum = 0.f;
#pragma unroll
  for (int k = 0; k < 8; ++k) sum += __expf(lp[k] - mx);
  float lse = mx + __logf(sum);
  float xmix = 0.f, lpd[8], lpm = -1e30f;
#pragma unroll
  for (int k = 0; k < 8; ++k) {
    lp[k] -= lse;
    float muk = pp[60 + k * 6 + c0];
    float sk = pp[108 + k * 6 + c0];
    float u = (z0v - muk) * __expf(-sk);
    xmix += __expf(lp[k]) * sigmf(u);
    lpd[k] = lp[k] - sk + logsigf(u) + logsigf(-u);
    lpm = fmaxf(lpm, lpd[k]);
  }
  float s2 = 0.f;
#pragma unroll
  for (int k = 0; k < 8; ++k) s2 += __expf(lpd[k] - lpm);
  float log_pdf = lpm + __logf(s2);
  float xc = fminf(fmaxf(xmix, 1e-6f), 1.f - 1e-6f);
  float lx = __logf(xc), l1mx = log1pf(-xc);
  float z0n = (lx - l1mx) * __expf(pa) + pb;
  float ldd = log_pdf + (-lx - l1mx) + pa;

  int yh = t >> 5, xh = t & 31;
  int c = (c0 < 3) ? c0 : c0 - 3;
  float* ob = out + ((size_t)b * 3 + c) * 4096;
  float z1v = z1[(size_t)b * 6144 + (size_t)c0 * 1024 + t];
  if (c0 < 3) {
    ob[(2 * yh) * 64 + 2 * xh] = z0n;       // p00
    ob[(2 * yh) * 64 + 2 * xh + 1] = z1v;   // p01
  } else {
    ob[(2 * yh + 1) * 64 + 2 * xh + 1] = z0n;  // p11
    ob[(2 * yh + 1) * 64 + 2 * xh] = z1v;      // p10
  }
#pragma unroll
  for (int off = 16; off > 0; off >>= 1) ldd += __shfl_xor(ldd, off, 32);
  if ((threadIdx.x & 31) == 0) atomicAdd(&outld[b], ldd);
}

// ---------------------------------------------------------------------------
extern "C" void kernel_launch(void* const* d_in, const int* in_sizes, int n_in,
                              void* d_out, int out_size, void* d_ws, size_t ws_size,
                              hipStream_t stream) {
  const float* z        = (const float*)d_in[0];
  const float* ld_in    = (const float*)d_in[1];
  const float* conv1_w  = (const float*)d_in[2];
  const float* conv1_b  = (const float*)d_in[3];
  const float* gconv_w  = (const float*)d_in[4];
  const float* gconv_b  = (const float*)d_in[5];
  const float* ln1_g    = (const float*)d_in[6];
  const float* ln1_b    = (const float*)d_in[7];
  const float* qkv_w    = (const float*)d_in[8];
  const float* qkv_b    = (const float*)d_in[9];
  const float* proj_w   = (const float*)d_in[10];
  const float* proj_b   = (const float*)d_in[11];
  const float* ln2_g    = (const float*)d_in[12];
  const float* ln2_b    = (const float*)d_in[13];
  const float* conv2_w  = (const float*)d_in[14];
  const float* conv2_b  = (const float*)d_in[15];
  const float* a_ls     = (const float*)d_in[16];
  const float* a_bs     = (const float*)d_in[17];

  unsigned char* ws = (unsigned char*)d_ws;
  size_t off = 0;
  auto alloc = [&](size_t bytes) {
    size_t r = off;
    off += (bytes + 255) & ~(size_t)255;
    return r;
  };
  const size_t TT = (size_t)NB * TOK;  // 65536 tokens
  float*     z0    = (float*)(ws + alloc((size_t)NB * 6144 * 4));
  float*     z1    = (float*)(ws + alloc((size_t)NB * 6144 * 4));
  float*     x32   = (float*)(ws + alloc(TT * FC * 4));   // conv1 out, later attn-gate out
  _Float16*  xh    = (_Float16*)(ws + alloc(TT * FC * 2)); // conv1 f16, later ln2 f16
  float*     gbuf  = (float*)(ws + alloc(TT * 192 * 4));   // gconv/proj out, later params(160)
  float*     x2    = (float*)(ws + alloc(TT * FC * 4));
  float*     ln1o  = (float*)(ws + alloc(TT * FC * 4));
  _Float16*  ln1h  = (_Float16*)(ws + alloc(TT * FC * 2));
  _Float16*  qkvh  = (_Float16*)(ws + alloc(TT * 288 * 2));
  _Float16*  oh    = (_Float16*)(ws + alloc(TT * FC * 2));
  _Float16*  pk_gc = (_Float16*)(ws + alloc((size_t)9 * 3 * 12 * 512 * 2));
  _Float16*  pk_qk = (_Float16*)(ws + alloc((size_t)3 * 18 * 512 * 2));
  _Float16*  pk_pr = (_Float16*)(ws + alloc((size_t)3 * 12 * 512 * 2));
  _Float16*  pk_c2 = (_Float16*)(ws + alloc((size_t)9 * 3 * 10 * 512 * 2));
  float*     stats = (float*)(ws + alloc(4 * NB * 4));  // ln1[128], ln2[128]
  float*     stats2 = stats + 2 * NB;

  float* out   = (float*)d_out;
  float* outld = out + (size_t)NB * 3 * 64 * 64;

  init_kernel<<<1, 256, 0, stream>>>(ld_in, outld, stats);

  // weight packing (B-fragment-major f16)
  pack_kernel<<<648, 256, 0, stream>>>(gconv_w, pk_gc, 3, 12, 192, 96 * 9, 9, 9);
  pack_kernel<<<108, 256, 0, stream>>>(qkv_w,  pk_qk, 3, 18, 288, 96, 1, 1);
  pack_kernel<<<72,  256, 0, stream>>>(proj_w, pk_pr, 3, 12, 192, 96, 1, 1);
  pack_kernel<<<540, 256, 0, stream>>>(conv2_w, pk_c2, 3, 10, 156, 96 * 9, 9, 9);

  split_kernel<<<768, 256, 0, stream>>>(z, z0, z1);

  conv1_kernel<<<dim3(384, NB), 256, 0, stream>>>(z1, conv1_w, conv1_b, x32, xh);

  // gconv (96 -> 192) via WMMA tap GEMMs
  convtap_kernel<<<dim3(6, 32, NB), 128, 0, stream>>>(xh, pk_gc, gconv_b, gbuf, 192,
                                                      12, 192);
  // gate 1 + LN1 stats
  gate_kernel<<<dim3(384, NB), 256, 0, stream>>>(x32, gbuf, x2, stats);
  ln_kernel<<<dim3(384, NB), 256, 0, stream>>>(x2, stats, ln1_g, ln1_b, ln1o, ln1h);

  // qkv GEMM (96 -> 288), f16 out
  gemm96_kernel<<<dim3(9, 32, NB), 128, 0, stream>>>(ln1h, pk_qk, qkv_b, 18, 288, 288,
                                                     nullptr, qkvh);
  // flash attention
  attn_kernel<<<dim3(16, NB), 128, 0, stream>>>(qkvh, oh);

  // proj GEMM (96 -> 192), f32 out into gbuf
  gemm96_kernel<<<dim3(6, 32, NB), 128, 0, stream>>>(oh, pk_pr, proj_b, 12, 192, 192,
                                                     gbuf, nullptr);
  // gate 2 (residual = ln1 output) + LN2 stats ; reuse x32
  gate_kernel<<<dim3(384, NB), 256, 0, stream>>>(ln1o, gbuf, x32, stats2);
  ln_kernel<<<dim3(384, NB), 256, 0, stream>>>(x32, stats2, ln2_g, ln2_b, nullptr, xh);

  // conv2 (96 -> 156, padded 160) via WMMA tap GEMMs, params into gbuf
  convtap_kernel<<<dim3(5, 32, NB), 128, 0, stream>>>(xh, pk_c2, conv2_b, gbuf, 160,
                                                      10, 156);
  // mixture-of-logistics + merge + log-det
  final_kernel<<<1536, 256, 0, stream>>>(z0, z1, gbuf, a_ls, a_bs, out, outld);
}
